// remoteclip_text_66675072303368
// MI455X (gfx1250) — matrix operand order repaired and checked by reading, not compiled, not run on hardware
//
#include <hip/hip_runtime.h>

typedef __attribute__((ext_vector_type(16))) _Float16 v16h;
typedef __attribute__((ext_vector_type(8)))  float    v8f;
typedef int i32x4 __attribute__((vector_size(16)));
typedef __attribute__((address_space(1))) i32x4 as1_i32x4;
typedef __attribute__((address_space(3))) i32x4 as3_i32x4;

#if __has_builtin(__builtin_amdgcn_global_load_async_to_lds_b128)
#define HAVE_ASYNC_LDS 1
#else
#define HAVE_ASYNC_LDS 0
#endif

struct __align__(16) U128 { unsigned u[4]; };

// 16B global -> LDS copy: async DMA (ASYNCcnt) when available, sync fallback.
__device__ __forceinline__ void copy16_g2l(const void* g, void* l) {
#if HAVE_ASYNC_LDS
  __builtin_amdgcn_global_load_async_to_lds_b128(
      (as1_i32x4*)(unsigned long long)g,
      (as3_i32x4*)(unsigned long long)l,
      0, 0);
#else
  *(U128*)l = *(const U128*)g;
#endif
}

template <int N> __device__ __forceinline__ void wait_async() {
#if __has_builtin(__builtin_amdgcn_s_wait_asynccnt)
  __builtin_amdgcn_s_wait_asynccnt(N);
#else
  asm volatile("s_wait_asynccnt %0" ::"i"(N) : "memory");
#endif
}

// ---------------------------------------------------------------------------
// Problem: B=2, C=2048, Hf=Wf=24, BLOCK=7, Ho=Wo=18, P=324 patches, T=50,
// heads=32, hd=64, OUT=512, NT=45. Only batch 1 reaches the output.
// ---------------------------------------------------------------------------

// Pack X(f16) = [ img(batch1) pixels (576 x 2048) ; pos_embed (50 x 2048) ;
//                 zero pad to 640 rows ], row-major, stride 2048.
__global__ void pack_xh(const float* __restrict__ img, const float* __restrict__ pe,
                        _Float16* __restrict__ X) {
  int idx = blockIdx.x * blockDim.x + threadIdx.x;       // 640*2048
  int p = idx >> 11, c = idx & 2047;
  float v = 0.f;
  if (p < 576)      v = img[(size_t)(2048 + c) * 576 + p];   // img[1, c, p]
  else if (p < 626) v = pe[(size_t)(p - 576) * 2048 + c];
  X[idx] = (_Float16)v;
}

// f32 -> f16 weight pack (done once; GEMM re-reads weights 3-5x from L2).
__global__ void pack_wh(const float* __restrict__ w, _Float16* __restrict__ h, int n) {
  int i = blockIdx.x * blockDim.x + threadIdx.x;
  if (i < n) h[i] = (_Float16)w[i];
}

// C(MxN,f32) = A(MxK,f16) * W(NxK,f16)^T via v_wmma_f32_16x16x32_f16.
// 256 threads = 8 wave32, C tile 128x64, K step 32, double-buffered LDS
// filled by global_load_async_to_lds_b128 (next tile overlaps current WMMAs).
__global__ __launch_bounds__(256) void gemm_wmma(
    const _Float16* __restrict__ A, const _Float16* __restrict__ W,
    float* __restrict__ C, int N, int K) {
  __shared__ _Float16 As[2][128 * 32];   // [m][k]
  __shared__ _Float16 Bs[2][64 * 32];    // [n][k]
  const int tid  = threadIdx.x;
  const int lane = tid & 31;
  const int wave = tid >> 5;
  const int lhi  = lane >> 4;
  const int llo  = lane & 15;
  const int m0 = blockIdx.y * 128;
  const int n0 = blockIdx.x * 64;

  v8f acc[4];
  v8f vz = {};
  acc[0] = vz; acc[1] = vz; acc[2] = vz; acc[3] = vz;

  // A: 2 threads/row, 16 halves (32B) each -> 2x b128 per thread
  const int arow = tid >> 1, acol = (tid & 1) * 16;
  // B: 4 threads/row, 8 halves (16B) each -> 1x b128 per thread
  const int brow = tid >> 2, bcol = (tid & 3) * 8;
  const _Float16* Ag = A + (size_t)(m0 + arow) * K + acol;
  const _Float16* Wg = W + (size_t)(n0 + brow) * K + bcol;
  _Float16* Asd0 = &As[0][arow * 32 + acol];
  _Float16* Bsd0 = &Bs[0][brow * 32 + bcol];
  _Float16* Asd1 = &As[1][arow * 32 + acol];
  _Float16* Bsd1 = &Bs[1][brow * 32 + bcol];

  const int nsteps = K >> 5;
  // prologue: tile 0 -> buffer 0
  copy16_g2l(Ag, Asd0);
  copy16_g2l(Ag + 8, Asd0 + 8);
  copy16_g2l(Wg, Bsd0);

  for (int i = 0; i < nsteps; ++i) {
    const int cur = i & 1;
    if (i + 1 < nsteps) {                 // issue next tile into other buffer
      const int k1 = (i + 1) << 5;
      _Float16* ad = cur ? Asd0 : Asd1;
      _Float16* bd = cur ? Bsd0 : Bsd1;
      copy16_g2l(Ag + k1, ad);
      copy16_g2l(Ag + k1 + 8, ad + 8);
      copy16_g2l(Wg + k1, bd);
      wait_async<3>();                    // tile i complete (in-order), 3 in flight
    } else {
      wait_async<0>();
    }
    __syncthreads();                      // all waves' DMA visible

    // A frag (16x32 f16): lanes 0-15 -> K 0..7 & 16..23; lanes 16-31 -> +8.
    union { unsigned u[8]; v16h h; } af, bf;
    const _Float16* Ar = &As[cur][(wave * 16 + llo) * 32];
#pragma unroll
    for (int j = 0; j < 8; ++j) {
      int ko = (j < 4) ? (lhi * 8 + 2 * j) : (16 + lhi * 8 + 2 * (j - 4));
      af.u[j] = *(const unsigned*)(Ar + ko);
    }
#pragma unroll
    for (int nt = 0; nt < 4; ++nt) {
      // B frag (32x16 f16): lanes 0-15 -> K 0..7 & 16..23; lanes 16-31 -> +8.
      const _Float16* Br = &Bs[cur][(nt * 16 + llo) * 32];
#pragma unroll
      for (int j = 0; j < 8; ++j) {
        int ko = (j < 4) ? (lhi * 8 + 2 * j) : (16 + lhi * 8 + 2 * (j - 4));
        bf.u[j] = *(const unsigned*)(Br + ko);
      }
      acc[nt] = __builtin_amdgcn_wmma_f32_16x16x32_f16(
          false, af.h, false, bf.h, (short)0, acc[nt], false, false);
    }
    __syncthreads();                      // reads done before buffer reuse
  }

  // C/D layout: VGPR r -> M = r + 8*lhi, N = llo
#pragma unroll
  for (int nt = 0; nt < 4; ++nt) {
    int col = n0 + nt * 16 + llo;
#pragma unroll
    for (int r = 0; r < 8; ++r) {
      int row = m0 + wave * 16 + lhi * 8 + r;
      C[(size_t)row * N + col] = acc[nt][r];
    }
  }
}

// Normalize the 45 text rows (512-d each).
__global__ void txt_norm(const float* __restrict__ txt, float* __restrict__ out) {
  __shared__ float red[128];
  int j = blockIdx.x, tid = threadIdx.x;
  float s = 0.f;
  for (int c = tid; c < 512; c += 128) { float v = txt[j * 512 + c]; s += v * v; }
  red[tid] = s; __syncthreads();
  for (int o = 64; o > 0; o >>= 1) { if (tid < o) red[tid] += red[tid + o]; __syncthreads(); }
  float inv = rsqrtf(fmaxf(red[0], 1e-24f));
  for (int c = tid; c < 512; c += 128) out[j * 512 + c] = txt[j * 512 + c] * inv;
}

// Per-patch attention pooling. One block per patch (324).
// G* are 640x2048 f32: rows 0..575 = per-pixel projections, 576..625 = pe proj.
__global__ __launch_bounds__(256) void attn_pool(
    const float* __restrict__ Gq, const float* __restrict__ Gk,
    const float* __restrict__ Gv, const float* __restrict__ qb,
    const float* __restrict__ kb, const float* __restrict__ vb,
    _Float16* __restrict__ Pooled) {
  __shared__ float q[2048], k0[2048], v0[2048];
  __shared__ float sc[32 * 50];
  __shared__ int   pr[49];
  int p = blockIdx.x;
  int oy = p / 18, ox = p % 18;
  int tid = threadIdx.x;
  if (tid < 49) { int di = tid / 7, dj = tid % 7; pr[tid] = (oy + di) * 24 + (ox + dj); }
  __syncthreads();

  const float inv49 = 1.f / 49.f;
  for (int c = tid; c < 2048; c += 256) {
    float sq = 0.f, sk = 0.f, sv = 0.f;
    for (int t = 0; t < 49; ++t) {
      size_t r = (size_t)pr[t] * 2048 + c;
      sq += Gq[r]; sk += Gk[r]; sv += Gv[r];
    }
    size_t pe0 = (size_t)576 * 2048 + c;
    q[c]  = sq * inv49 + Gq[pe0] + qb[c];
    k0[c] = sk * inv49 + Gk[pe0] + kb[c];
    v0[c] = sv * inv49 + Gv[pe0] + vb[c];
  }
  __syncthreads();

  // scores: 50 tokens x 32 heads, hd=64, scale 1/8
  for (int s = tid; s < 1600; s += 256) {
    int t = s / 32, h = s & 31;
    int cb = h * 64;
    float acc = 0.f;
    if (t == 0) {
      for (int d = 0; d < 64; ++d) acc += q[cb + d] * k0[cb + d];
    } else {
      const float* Kr = Gk + (size_t)pr[t - 1] * 2048 + cb;
      const float* Ke = Gk + (size_t)(576 + t) * 2048 + cb;
      for (int d = 0; d < 64; ++d) acc += q[cb + d] * (Kr[d] + Ke[d] + kb[cb + d]);
    }
    sc[h * 50 + t] = acc * 0.125f;
  }
  __syncthreads();

  if (tid < 32) {  // softmax per head
    int h = tid;
    float mx = -1e30f;
    for (int t = 0; t < 50; ++t) mx = fmaxf(mx, sc[h * 50 + t]);
    float sum = 0.f;
    for (int t = 0; t < 50; ++t) { float e = __expf(sc[h * 50 + t] - mx); sc[h * 50 + t] = e; sum += e; }
    float inv = 1.f / sum;
    for (int t = 0; t < 50; ++t) sc[h * 50 + t] *= inv;
  }
  __syncthreads();

  for (int c = tid; c < 2048; c += 256) {
    int h = c >> 6;
    const float* at = &sc[h * 50];
    float acc = at[0] * v0[c];
    for (int t = 1; t < 50; ++t)
      acc += at[t] * (Gv[(size_t)pr[t - 1] * 2048 + c] + Gv[(size_t)(576 + t) * 2048 + c] + vb[c]);
    Pooled[(size_t)p * 2048 + c] = (_Float16)acc;
  }
}

__global__ void zero_f16(_Float16* __restrict__ p, int n) {
  int i = blockIdx.x * blockDim.x + threadIdx.x;
  if (i < n) p[i] = (_Float16)0.f;
}

// logits vs normalized text, argmax -> bucket (searchsorted [8,15,22,29,36])
__global__ __launch_bounds__(64) void classify(
    const float* __restrict__ F, const float* __restrict__ ob,
    const float* __restrict__ Tn, int* __restrict__ bucket) {
  __shared__ float f[512];
  __shared__ float lg[45];
  int p = blockIdx.x, tid = threadIdx.x;
  for (int c = tid; c < 512; c += 64) f[c] = F[(size_t)p * 512 + c] + ob[c];
  __syncthreads();
  if (tid < 45) {
    float a = 0.f;
    for (int c = 0; c < 512; ++c) a += f[c] * Tn[tid * 512 + c];
    lg[tid] = a;   // row-scale of f is argmax-invariant; txt normalized already
  }
  __syncthreads();
  if (tid == 0) {
    int best = 0; float bv = lg[0];
    for (int j = 1; j < 45; ++j) if (lg[j] > bv) { bv = lg[j]; best = j; }
    bucket[p] = (best >= 8) + (best >= 15) + (best >= 22) + (best >= 29) + (best >= 36);
  }
}

// 7x7 box vote per pixel, argmax bucket; channel-norm collapses to
// 0 (bucket 0) or 1/sqrt(512).
__global__ __launch_bounds__(576) void votes_k(const int* __restrict__ bucket,
                                               float* __restrict__ val) {
  int t = threadIdx.x;
  int y = t / 24, x = t % 24;
  int cnt[6] = {0, 0, 0, 0, 0, 0};
  int oy0 = y - 6 < 0 ? 0 : y - 6, oy1 = y < 17 ? y : 17;
  int ox0 = x - 6 < 0 ? 0 : x - 6, ox1 = x < 17 ? x : 17;
  for (int oy = oy0; oy <= oy1; ++oy)
    for (int ox = ox0; ox <= ox1; ++ox)
      cnt[bucket[oy * 18 + ox]]++;
  int best = 0, bv = cnt[0];
#pragma unroll
  for (int k = 1; k < 6; ++k) if (cnt[k] > bv) { bv = cnt[k]; best = k; }
  val[t] = (best == 0) ? 0.f : 0.044194173824159216f;  // 1/sqrt(512)
}

__global__ void writeout(const float* __restrict__ val, float* __restrict__ out) {
  int i = blockIdx.x * blockDim.x + threadIdx.x;  // 2*512*576
  int b = i / (512 * 576);
  int pix = i % 576;
  out[i] = (b == 0) ? 0.f : val[pix];
}

extern "C" void kernel_launch(void* const* d_in, const int* in_sizes, int n_in,
                              void* d_out, int out_size, void* d_ws, size_t ws_size,
                              hipStream_t stream) {
  const float* img = (const float*)d_in[0];
  const float* txt = (const float*)d_in[1];
  const float* pe  = (const float*)d_in[2];
  const float* qw  = (const float*)d_in[3];
  const float* qb  = (const float*)d_in[4];
  const float* kw  = (const float*)d_in[5];
  const float* kb_ = (const float*)d_in[6];
  const float* vw  = (const float*)d_in[7];
  const float* vb  = (const float*)d_in[8];
  const float* ow  = (const float*)d_in[9];
  const float* ob  = (const float*)d_in[10];
  float* out = (float*)d_out;

  char* w = (char*)d_ws;                             // ~45.9 MiB total
  _Float16* Xh  = (_Float16*)(w + 0);                // 640x2048 f16
  _Float16* Wqh = (_Float16*)(w + 2621440);          // 2048x2048 f16
  _Float16* Wkh = (_Float16*)(w + 11010048);
  _Float16* Wvh = (_Float16*)(w + 19398656);
  _Float16* Woh = (_Float16*)(w + 27787264);         // 512x2048 f16
  float*    Gq  = (float*)(w + 29884416);            // 640x2048 f32
  float*    Gk  = (float*)(w + 35127296);
  float*    Gv  = (float*)(w + 40370176);
  _Float16* Pl  = (_Float16*)(w + 45613056);         // 384x2048 f16
  float*    F   = (float*)(w + 47185920);            // 384x512 f32
  float*    Tn  = (float*)(w + 47972352);            // 45x512 f32
  int*      Bk  = (int*)  (w + 48064512);            // 324 int
  float*    Vl  = (float*)(w + 48066560);            // 576 f32

  pack_xh<<<640 * 2048 / 256, 256, 0, stream>>>(img, pe, Xh);
  pack_wh<<<2048 * 2048 / 256, 256, 0, stream>>>(qw, Wqh, 2048 * 2048);
  pack_wh<<<2048 * 2048 / 256, 256, 0, stream>>>(kw, Wkh, 2048 * 2048);
  pack_wh<<<2048 * 2048 / 256, 256, 0, stream>>>(vw, Wvh, 2048 * 2048);
  pack_wh<<<512 * 2048 / 256, 256, 0, stream>>>(ow, Woh, 512 * 2048);

  dim3 g1(2048 / 64, 640 / 128);                     // (N tiles, M tiles)
  gemm_wmma<<<g1, 256, 0, stream>>>(Xh, Wqh, Gq, 2048, 2048);
  gemm_wmma<<<g1, 256, 0, stream>>>(Xh, Wkh, Gk, 2048, 2048);
  gemm_wmma<<<g1, 256, 0, stream>>>(Xh, Wvh, Gv, 2048, 2048);

  txt_norm<<<45, 128, 0, stream>>>(txt, Tn);

  attn_pool<<<324, 256, 0, stream>>>(Gq, Gk, Gv, qb, kb_, vb, Pl);
  zero_f16<<<(60 * 2048 + 255) / 256, 256, 0, stream>>>(Pl + (size_t)324 * 2048, 60 * 2048);

  dim3 g2(512 / 64, 384 / 128);
  gemm_wmma<<<g2, 256, 0, stream>>>(Pl, Woh, F, 512, 2048);

  classify<<<324, 64, 0, stream>>>(F, ob, Tn, Bk);
  votes_k<<<1, 576, 0, stream>>>(Bk, Vl);
  writeout<<<(2 * 512 * 576) / 256, 256, 0, stream>>>(Vl, out);
}
